// MambaDiffusionBlock_64656437674689
// MI455X (gfx1250) — matrix operand order repaired and checked by reading, not compiled, not run on hardware
//
#include <hip/hip_runtime.h>
#include <hip/hip_bf16.h>
#include <math.h>

// ---------------------------------------------------------------------------
// MambaDiffusionBlock for MI455X (gfx1250): bf16 WMMA GEMMs + VALU scan.
// Pipelined GEMM: register-prefetch + double-buffered LDS, 1 barrier / slab.
// ---------------------------------------------------------------------------

typedef __attribute__((ext_vector_type(16))) __bf16 v16bf;
typedef __attribute__((ext_vector_type(8)))  __bf16 v8bf;
typedef __attribute__((ext_vector_type(8)))  float  v8f;

#define DIMC     512
#define D_INNER  1024
#define D_STATE  16
#define DT_RANK  32
#define N_HEADS  8
#define HEAD_D   64
#define BATCH    4
#define SEQ_L    1024
#define TXT_L    77
#define M_ROWS   (BATCH * SEQ_L)     // 4096
#define KV_ROWS  (BATCH * TXT_L)     // 308

#define ACT_NONE     0
#define ACT_SOFTPLUS 1
#define ACT_GELU     2

#define LDS_STRIDE   40              // bf16 elems; 80B rows: 16B-aligned, bank-friendly

// ---------------------------------------------------------------------------
// LayerNorm: one 256-thread block per row (cols = 512).
// ---------------------------------------------------------------------------
__global__ void __launch_bounds__(256)
layernorm_kernel(const float* __restrict__ x, const float* __restrict__ w,
                 const float* __restrict__ b, float* __restrict__ out, int cols)
{
    const int row = blockIdx.x;
    const float* xr = x + (size_t)row * cols;
    float* orow = out + (size_t)row * cols;
    __shared__ float red1[256];
    __shared__ float red2[256];
    float s = 0.f, s2 = 0.f;
    for (int c = threadIdx.x; c < cols; c += 256) {
        float v = xr[c];
        s += v; s2 += v * v;
    }
    red1[threadIdx.x] = s;
    red2[threadIdx.x] = s2;
    __syncthreads();
    for (int off = 128; off > 0; off >>= 1) {
        if (threadIdx.x < off) {
            red1[threadIdx.x] += red1[threadIdx.x + off];
            red2[threadIdx.x] += red2[threadIdx.x + off];
        }
        __syncthreads();
    }
    float mu  = red1[0] / (float)cols;
    float var = red2[0] / (float)cols - mu * mu;
    float inv = rsqrtf(var + 1e-5f);
    for (int c = threadIdx.x; c < cols; c += 256)
        orow[c] = (xr[c] - mu) * inv * w[c] + b[c];
}

// ---------------------------------------------------------------------------
// Fragment loaders (ISA 7.12.2 layouts) — two 16B ds_load_b128 each.
// ---------------------------------------------------------------------------
__device__ __forceinline__ v16bf frag_a(const __bf16* rowp, int hf)
{
    v8bf lo = *(const v8bf*)(rowp + hf * 8);
    v8bf hi = *(const v8bf*)(rowp + 16 + hf * 8);
    return __builtin_shufflevector(lo, hi, 0,1,2,3,4,5,6,7,8,9,10,11,12,13,14,15);
}
__device__ __forceinline__ v16bf frag_b(const __bf16* rowp, int hf)
{
    v8bf lo = *(const v8bf*)(rowp + hf * 8);
    v8bf hi = *(const v8bf*)(rowp + 16 + hf * 8);
    return __builtin_shufflevector(lo, hi, 0,1,2,3,4,5,6,7,8,9,10,11,12,13,14,15);
}
__device__ __forceinline__ v8bf pack_bf8(float4 f0, float4 f1)
{
    v8bf pk;
    pk[0] = (__bf16)f0.x; pk[1] = (__bf16)f0.y;
    pk[2] = (__bf16)f0.z; pk[3] = (__bf16)f0.w;
    pk[4] = (__bf16)f1.x; pk[5] = (__bf16)f1.y;
    pk[6] = (__bf16)f1.z; pk[7] = (__bf16)f1.w;
    return pk;
}

// One thread's share of a 32-wide K slab: 2 A chunks + 1 W chunk (8 f32 each).
struct Slab {
    float4 a00, a01, a10, a11, w0, w1;
};

// ---------------------------------------------------------------------------
// bf16 WMMA GEMM:  out[M,N] = epilogue( A[M,K](lda) @ W[N,K]^T )
//   - 128x64 tile per 256-thread block (8 waves); wave owns 32x32 sub-tile
//   - 4 x v_wmma_f32_16x16x32_bf16 per K-slab, software-pipelined:
//     regs hold slab k+1 while LDS buffer b holds slab k (ping-pong).
// ---------------------------------------------------------------------------
template<int ACT>
__global__ void __launch_bounds__(256)
gemm_bf16_wmma(const float* __restrict__ A, const float* __restrict__ W,
               const float* __restrict__ bias,
               const float* __restrict__ res1, const float* __restrict__ res2,
               float* __restrict__ out,
               int M, int N, int K, int lda)
{
    __shared__ __align__(16) __bf16 As[2][128][LDS_STRIDE];
    __shared__ __align__(16) __bf16 Ws[2][64][LDS_STRIDE];

    const int tid  = threadIdx.x;
    const int lane = tid & 31;
    const int wave = tid >> 5;
    const int wm   = wave >> 1;     // 0..3  -> M sub-tile (32 rows each)
    const int wn   = wave & 1;      // 0..1  -> N sub-tile (32 cols each)
    const int hf   = lane >> 4;     // lane half (ISA 7.12.2)
    const int lr   = lane & 15;
    const int m0   = blockIdx.y * 128;
    const int n0   = blockIdx.x * 64;

    // staging chunk decode: 8 bf16 per chunk, 4 chunks per 32-wide row
    const int rA = tid >> 2;              // 0..63  (A uses rA and rA+64)
    const int cc = (tid & 3) * 8;         // 0,8,16,24

    // clamped global row pointers (row-OOB only feeds unstored outputs)
    int gm0 = m0 + rA;        if (gm0 > M - 1) gm0 = M - 1;
    int gm1 = m0 + rA + 64;   if (gm1 > M - 1) gm1 = M - 1;
    const float* pa0 = A + (size_t)gm0 * lda + cc;
    const float* pa1 = A + (size_t)gm1 * lda + cc;
    const float* pw  = W + (size_t)(n0 + rA) * K + cc;

    v8f c00 = {}; v8f c01 = {}; v8f c10 = {}; v8f c11 = {};

    auto load_slab = [&](int kt, Slab& s) {
        s.a00 = *(const float4*)(pa0 + kt);
        s.a01 = *(const float4*)(pa0 + kt + 4);
        s.a10 = *(const float4*)(pa1 + kt);
        s.a11 = *(const float4*)(pa1 + kt + 4);
        s.w0  = *(const float4*)(pw + kt);
        s.w1  = *(const float4*)(pw + kt + 4);
    };
    auto store_slab = [&](const Slab& s, int buf) {
        *(v8bf*)&As[buf][rA][cc]      = pack_bf8(s.a00, s.a01);
        *(v8bf*)&As[buf][rA + 64][cc] = pack_bf8(s.a10, s.a11);
        *(v8bf*)&Ws[buf][rA][cc]      = pack_bf8(s.w0, s.w1);
    };
    auto compute = [&](int buf) {
        v16bf a0 = frag_a(&As[buf][wm * 32 + lr][0],      hf);
        v16bf a1 = frag_a(&As[buf][wm * 32 + 16 + lr][0], hf);
        v16bf b0 = frag_b(&Ws[buf][wn * 32 + lr][0],      hf);
        v16bf b1 = frag_b(&Ws[buf][wn * 32 + 16 + lr][0], hf);
        c00 = __builtin_amdgcn_wmma_f32_16x16x32_bf16(false, a0, false, b0,
                                                      (short)0, c00, false, false);
        c01 = __builtin_amdgcn_wmma_f32_16x16x32_bf16(false, a0, false, b1,
                                                      (short)0, c01, false, false);
        c10 = __builtin_amdgcn_wmma_f32_16x16x32_bf16(false, a1, false, b0,
                                                      (short)0, c10, false, false);
        c11 = __builtin_amdgcn_wmma_f32_16x16x32_bf16(false, a1, false, b1,
                                                      (short)0, c11, false, false);
    };

    if (K == 32) {
        // Single-slab path (dt_proj).
        Slab s;
        load_slab(0, s);
        store_slab(s, 0);
        __syncthreads();
        compute(0);
    } else {
        // Pipelined path: K % 64 == 0 at all remaining call sites.
        Slab s0, s1;
        load_slab(0, s0);
        for (int kt = 0; kt < K; kt += 64) {
            store_slab(s0, 0);
            load_slab(kt + 32, s1);              // in flight during compute(0)
            if (kt + 96 < K) {
                // CDNA5 speculative prefetch two slabs ahead (global_prefetch_b8)
                __builtin_prefetch(pa0 + kt + 96, 0, 1);
            }
            __syncthreads();
            compute(0);

            store_slab(s1, 1);                   // waits loadcnt here, post-WMMA
            if (kt + 64 < K) load_slab(kt + 64, s0);
            __syncthreads();
            compute(1);
        }
    }

    // Epilogue. C/D layout: M = hf*8 + r, N = lr within each 16x16 tile.
    v8f accs[2][2] = { { c00, c01 }, { c10, c11 } };
    #pragma unroll
    for (int mi = 0; mi < 2; mi++) {
        #pragma unroll
        for (int ni = 0; ni < 2; ni++) {
            v8f cv = accs[mi][ni];
            const int n = n0 + wn * 32 + ni * 16 + lr;
            #pragma unroll
            for (int r = 0; r < 8; r++) {
                int m = m0 + wm * 32 + mi * 16 + hf * 8 + r;
                if (m < M) {
                    float val = cv[r];
                    if (bias) val += bias[n];
                    if (ACT == ACT_SOFTPLUS) {
                        val = (val > 20.f) ? val : log1pf(__expf(val));
                    } else if (ACT == ACT_GELU) {
                        val = 0.5f * val * (1.f + erff(val * 0.70710678118654752f));
                    }
                    size_t o = (size_t)m * N + n;
                    if (res1) val += res1[o];
                    if (res2) val += res2[o];
                    out[o] = val;
                }
            }
        }
    }
}

// ---------------------------------------------------------------------------
// Depthwise causal conv (4 taps) + SiLU. Reads xc-half of xz [M, 2048].
// ---------------------------------------------------------------------------
__global__ void __launch_bounds__(256)
conv_silu_kernel(const float* __restrict__ xz, const float* __restrict__ cw,
                 const float* __restrict__ cb, float* __restrict__ xc,
                 int total, int L)
{
    int idx = blockIdx.x * blockDim.x + threadIdx.x;
    if (idx >= total) return;
    int d    = idx & (D_INNER - 1);
    int row  = idx >> 10;
    int l    = row % L;
    int rowb = row - l;
    float acc = cb[d];
    #pragma unroll
    for (int k = 0; k < 4; k++) {
        int ls = l + k - 3;
        if (ls >= 0)
            acc += xz[(size_t)(rowb + ls) * (2 * D_INNER) + d] * cw[d * 4 + k];
    }
    xc[idx] = acc / (1.f + __expf(-acc));   // silu
}

// ---------------------------------------------------------------------------
// Selective scan. grid = (D_INNER/256, B), block = 256. Thread owns (b, d)
// and carries its 16 states in registers; B/C vectors staged in LDS per step.
// Fuses  y = (sum_n h_n*C_n + D*xc) * silu(z).
// ---------------------------------------------------------------------------
__global__ void __launch_bounds__(256)
scan_kernel(const float* __restrict__ xz,   // [M, 2048]   (z half)
            const float* __restrict__ xc,   // [M, 1024]
            const float* __restrict__ dt,   // [M, 1024]
            const float* __restrict__ dbl,  // [M, 64]     (B: 32..47, C: 48..63)
            const float* __restrict__ A_log,// [1024, 16]
            const float* __restrict__ Dp,   // [1024]
            float* __restrict__ ys,         // [M, 1024]
            int L)
{
    const int b = blockIdx.y;
    const int d = blockIdx.x * 256 + threadIdx.x;

    float Aa[D_STATE];
    #pragma unroll
    for (int n = 0; n < D_STATE; n++) Aa[n] = -__expf(A_log[d * D_STATE + n]);
    float h[D_STATE];
    #pragma unroll
    for (int n = 0; n < D_STATE; n++) h[n] = 0.f;
    const float Dd = Dp[d];

    __shared__ float Bs[D_STATE];
    __shared__ float Cs[D_STATE];

    for (int l = 0; l < L; l++) {
        size_t row = (size_t)b * L + l;
        if (threadIdx.x < 2 * D_STATE) {
            float vv = dbl[row * 64 + DT_RANK + threadIdx.x];
            if (threadIdx.x < D_STATE) Bs[threadIdx.x] = vv;
            else                       Cs[threadIdx.x - D_STATE] = vv;
        }
        __syncthreads();
        float dtv = dt[row * D_INNER + d];
        float xv  = xc[row * D_INNER + d];
        float y = 0.f;
        #pragma unroll
        for (int n = 0; n < D_STATE; n++) {
            float dA = __expf(dtv * Aa[n]);
            h[n] = dA * h[n] + dtv * Bs[n] * xv;
            y += h[n] * Cs[n];
        }
        y += Dd * xv;
        float z  = xz[row * (2 * D_INNER) + D_INNER + d];
        float sz = z / (1.f + __expf(-z));
        ys[row * D_INNER + d] = y * sz;
        __syncthreads();
    }
}

// ---------------------------------------------------------------------------
// Cross-attention core (77 keys). block = 64 query threads for one (b, h);
// K, V and per-thread scores staged in LDS (< 64 KB).
// ---------------------------------------------------------------------------
__global__ void __launch_bounds__(64)
attn_kernel(const float* __restrict__ q,  // [M, 512]
            const float* __restrict__ k,  // [308, 512]
            const float* __restrict__ v,  // [308, 512]
            float* __restrict__ out,      // [M, 512]
            int Lq, int Lk)
{
    const int b = blockIdx.z;
    const int hh = blockIdx.y;
    const int qi = blockIdx.x * 64 + threadIdx.x;

    __shared__ float Ks[TXT_L][HEAD_D + 1];
    __shared__ float Vs[TXT_L][HEAD_D + 1];
    __shared__ float Sc[64][TXT_L + 1];

    for (int i = threadIdx.x; i < Lk * HEAD_D; i += 64) {
        int t = i / HEAD_D, dd = i % HEAD_D;
        size_t src = ((size_t)b * Lk + t) * DIMC + hh * HEAD_D + dd;
        Ks[t][dd] = k[src];
        Vs[t][dd] = v[src];
    }
    __syncthreads();

    size_t qrow = (size_t)b * Lq + qi;
    float qv[HEAD_D];
    #pragma unroll
    for (int dd = 0; dd < HEAD_D; dd++)
        qv[dd] = q[qrow * DIMC + hh * HEAD_D + dd];

    float mx = -3.4e38f;
    for (int t = 0; t < Lk; t++) {
        float s = 0.f;
        #pragma unroll
        for (int dd = 0; dd < HEAD_D; dd++) s += qv[dd] * Ks[t][dd];
        s *= 0.125f;                         // 1/sqrt(64)
        Sc[threadIdx.x][t] = s;
        mx = fmaxf(mx, s);
    }
    float den = 0.f;
    for (int t = 0; t < Lk; t++) {
        float e = __expf(Sc[threadIdx.x][t] - mx);
        Sc[threadIdx.x][t] = e;
        den += e;
    }
    float inv = 1.f / den;
    float acc[HEAD_D];
    #pragma unroll
    for (int dd = 0; dd < HEAD_D; dd++) acc[dd] = 0.f;
    for (int t = 0; t < Lk; t++) {
        float wgt = Sc[threadIdx.x][t] * inv;
        #pragma unroll
        for (int dd = 0; dd < HEAD_D; dd++) acc[dd] += wgt * Vs[t][dd];
    }
    #pragma unroll
    for (int dd = 0; dd < HEAD_D; dd++)
        out[qrow * DIMC + hh * HEAD_D + dd] = acc[dd];
}

// ---------------------------------------------------------------------------
// Host-side orchestration.
// ---------------------------------------------------------------------------
extern "C" void kernel_launch(void* const* d_in, const int* in_sizes, int n_in,
                              void* d_out, int out_size, void* d_ws, size_t ws_size,
                              hipStream_t stream)
{
    const float* x        = (const float*)d_in[0];
    const float* text     = (const float*)d_in[1];
    const float* ln1_w    = (const float*)d_in[2];
    const float* ln1_b    = (const float*)d_in[3];
    const float* ln2_w    = (const float*)d_in[4];
    const float* ln2_b    = (const float*)d_in[5];
    const float* ln3_w    = (const float*)d_in[6];
    const float* ln3_b    = (const float*)d_in[7];
    const float* in_proj  = (const float*)d_in[8];
    const float* conv_w   = (const float*)d_in[9];
    const float* conv_b   = (const float*)d_in[10];
    const float* x_proj   = (const float*)d_in[11];
    const float* dt_proj  = (const float*)d_in[12];
    const float* dt_projb = (const float*)d_in[13];
    const float* A_log    = (const float*)d_in[14];
    const float* Dp       = (const float*)d_in[15];
    const float* out_proj = (const float*)d_in[16];
    const float* attn_inw = (const float*)d_in[17];
    const float* attn_inb = (const float*)d_in[18];
    const float* attn_ow  = (const float*)d_in[19];
    const float* attn_ob  = (const float*)d_in[20];
    const float* ffn_w1   = (const float*)d_in[21];
    const float* ffn_b1   = (const float*)d_in[22];
    const float* ffn_w2   = (const float*)d_in[23];
    const float* ffn_b2   = (const float*)d_in[24];
    float* outp = (float*)d_out;
    float* ws   = (float*)d_ws;

    // Workspace layout (float offsets), with lifetime-based aliasing.
    const size_t OFF_XZ  = 0;                                        // [4096,2048] (later: ffn hidden)
    const size_t OFF_XC  = OFF_XZ  + (size_t)M_ROWS * 2 * D_INNER;   // [4096,1024]
    const size_t OFF_DT  = OFF_XC  + (size_t)M_ROWS * D_INNER;       // [4096,1024]
    const size_t OFF_YS  = OFF_DT  + (size_t)M_ROWS * D_INNER;       // [4096,1024] (later: g)
    const size_t OFF_H   = OFF_YS  + (size_t)M_ROWS * D_INNER;       // [4096,512] ln out
    const size_t OFF_X1  = OFF_H   + (size_t)M_ROWS * DIMC;          // [4096,512]
    const size_t OFF_X2  = OFF_X1  + (size_t)M_ROWS * DIMC;          // [4096,512]
    const size_t OFF_DBL = OFF_X2  + (size_t)M_ROWS * DIMC;          // [4096,64]
    // aliases (dead-by-then regions):
    const size_t OFF_Q   = OFF_XC;                                   // [4096,512]
    const size_t OFF_AP  = OFF_XC + (size_t)M_ROWS * DIMC;           // [4096,512]
    const size_t OFF_K   = OFF_DT;                                   // [308,512]
    const size_t OFF_V   = OFF_DT + (size_t)KV_ROWS * DIMC;          // [308,512]
    const size_t OFF_G   = OFF_YS;                                   // [4096,512]

    dim3 blk256(256);

    // 1) h1 = LN1(x)
    layernorm_kernel<<<M_ROWS, blk256, 0, stream>>>(x, ln1_w, ln1_b, ws + OFF_H, DIMC);

    // 2) xz = h1 @ in_proj^T       [4096,2048]
    gemm_bf16_wmma<ACT_NONE><<<dim3(2 * D_INNER / 64, M_ROWS / 128), blk256, 0, stream>>>(
        ws + OFF_H, in_proj, nullptr, nullptr, nullptr, ws + OFF_XZ,
        M_ROWS, 2 * D_INNER, DIMC, DIMC);

    // 3) xc = silu(conv(xz[:, :1024]))
    conv_silu_kernel<<<(M_ROWS * D_INNER) / 256, blk256, 0, stream>>>(
        ws + OFF_XZ, conv_w, conv_b, ws + OFF_XC, M_ROWS * D_INNER, SEQ_L);

    // 4) dbl = xc @ x_proj^T       [4096,64]
    gemm_bf16_wmma<ACT_NONE><<<dim3(1, M_ROWS / 128), blk256, 0, stream>>>(
        ws + OFF_XC, x_proj, nullptr, nullptr, nullptr, ws + OFF_DBL,
        M_ROWS, 64, D_INNER, D_INNER);

    // 5) dt = softplus(dbl[:, :32] @ dt_proj^T + b)   [4096,1024]
    gemm_bf16_wmma<ACT_SOFTPLUS><<<dim3(D_INNER / 64, M_ROWS / 128), blk256, 0, stream>>>(
        ws + OFF_DBL, dt_proj, dt_projb, nullptr, nullptr, ws + OFF_DT,
        M_ROWS, D_INNER, DT_RANK, 64);

    // 6) selective scan -> ys (includes D*xc and *silu(z))
    scan_kernel<<<dim3(D_INNER / 256, BATCH), blk256, 0, stream>>>(
        ws + OFF_XZ, ws + OFF_XC, ws + OFF_DT, ws + OFF_DBL, A_log, Dp,
        ws + OFF_YS, SEQ_L);

    // 7) x1 = ys @ out_proj^T + x  [4096,512]
    gemm_bf16_wmma<ACT_NONE><<<dim3(DIMC / 64, M_ROWS / 128), blk256, 0, stream>>>(
        ws + OFF_YS, out_proj, nullptr, x, nullptr, ws + OFF_X1,
        M_ROWS, DIMC, D_INNER, D_INNER);

    // 8) h2 = LN2(x1)
    layernorm_kernel<<<M_ROWS, blk256, 0, stream>>>(ws + OFF_X1, ln2_w, ln2_b, ws + OFF_H, DIMC);

    // 9) q = h2 @ wq^T + bq ; k,v from text_emb
    gemm_bf16_wmma<ACT_NONE><<<dim3(DIMC / 64, M_ROWS / 128), blk256, 0, stream>>>(
        ws + OFF_H, attn_inw, attn_inb, nullptr, nullptr, ws + OFF_Q,
        M_ROWS, DIMC, DIMC, DIMC);
    gemm_bf16_wmma<ACT_NONE><<<dim3(DIMC / 64, (KV_ROWS + 127) / 128), blk256, 0, stream>>>(
        text, attn_inw + (size_t)DIMC * DIMC, attn_inb + DIMC, nullptr, nullptr, ws + OFF_K,
        KV_ROWS, DIMC, DIMC, DIMC);
    gemm_bf16_wmma<ACT_NONE><<<dim3(DIMC / 64, (KV_ROWS + 127) / 128), blk256, 0, stream>>>(
        text, attn_inw + (size_t)2 * DIMC * DIMC, attn_inb + 2 * DIMC, nullptr, nullptr, ws + OFF_V,
        KV_ROWS, DIMC, DIMC, DIMC);

    // 10) attention core -> attn_pre
    attn_kernel<<<dim3(SEQ_L / 64, N_HEADS, BATCH), dim3(64), 0, stream>>>(
        ws + OFF_Q, ws + OFF_K, ws + OFF_V, ws + OFF_AP, SEQ_L, TXT_L);

    // 11) x2 = attn_pre @ attn_out^T + b + h2 + x1
    gemm_bf16_wmma<ACT_NONE><<<dim3(DIMC / 64, M_ROWS / 128), blk256, 0, stream>>>(
        ws + OFF_AP, attn_ow, attn_ob, ws + OFF_H, ws + OFF_X1, ws + OFF_X2,
        M_ROWS, DIMC, DIMC, DIMC);

    // 12) g = LN3(x2)
    layernorm_kernel<<<M_ROWS, blk256, 0, stream>>>(ws + OFF_X2, ln3_w, ln3_b, ws + OFF_G, DIMC);

    // 13) f1 = gelu(g @ ffn_w1^T + b1)   [4096,2048]  (reuses XZ region)
    gemm_bf16_wmma<ACT_GELU><<<dim3(4 * DIMC / 64, M_ROWS / 128), blk256, 0, stream>>>(
        ws + OFF_G, ffn_w1, ffn_b1, nullptr, nullptr, ws + OFF_XZ,
        M_ROWS, 4 * DIMC, DIMC, DIMC);

    // 14) out = f1 @ ffn_w2^T + b2 + x2
    gemm_bf16_wmma<ACT_NONE><<<dim3(DIMC / 64, M_ROWS / 128), blk256, 0, stream>>>(
        ws + OFF_XZ, ffn_w2, ffn_b2, ws + OFF_X2, nullptr, outp,
        M_ROWS, DIMC, 4 * DIMC, 4 * DIMC);
}
